// PairwiseRelationNetwork_20074677141790
// MI455X (gfx1250) — matrix operand order repaired and checked by reading, not compiled, not run on hardware
//
#include <hip/hip_runtime.h>

typedef __attribute__((ext_vector_type(16))) _Float16 v16h;
typedef __attribute__((ext_vector_type(8)))  _Float16 v8h;
typedef __attribute__((ext_vector_type(4)))  _Float16 v4h;
typedef __attribute__((ext_vector_type(8)))  float    v8f;

union V16U { v16h v; v8h h[2]; };

#define XS_STRIDE 136   // 16x128 f16 x-tile, padded to avoid bank conflicts
#define HS_STRIDE 136   // 128x128 f16 activation buffer, padded
#define NTHREADS  512   // 16 waves: 8 M-tiles x 2 N-halves

// ---------------------------------------------------------------------------
// Kernel 0: convert f32 weights -> f16, pre-swizzled into WMMA B-fragment
// layout. For layer l, fragment (kt,nt) is 512 contiguous halfs:
//   wp[lbase + ((kt*8+nt)<<9) + lane*16 + e] = W[(kt*32 + (lane>>4)*16 + e)*128
//                                               + nt*16 + (lane&15)]
// L1: 256x128 (32768 halfs) | L2: 128x128 (16384) | L3: 128x128 (16384)
// ---------------------------------------------------------------------------
__global__ __launch_bounds__(256) void prn_pack_weights(
    const float* __restrict__ W1, const float* __restrict__ W2,
    const float* __restrict__ W3, _Float16* __restrict__ wp)
{
  int tid = blockIdx.x * 256 + threadIdx.x;   // 0..65535
  const float* W; int loff, lbase;
  if (tid < 32768)      { W = W1; loff = tid;         lbase = 0;     }
  else if (tid < 49152) { W = W2; loff = tid - 32768; lbase = 32768; }
  else                  { W = W3; loff = tid - 49152; lbase = 49152; }
  int e    = loff & 15;
  int lane = (loff >> 4) & 31;
  int frag = loff >> 9;
  int nt   = frag & 7;
  int kt   = frag >> 3;
  int col  = nt * 16 + (lane & 15);
  int K    = kt * 32 + (lane >> 4) * 16 + e;
  wp[lbase + loff] = (_Float16)W[K * 128 + col];
}

// LDS byte offset of a pointer into a __shared__ array (addrspacecast -> AS3).
__device__ __forceinline__ unsigned lds_off(const void* p) {
  return (unsigned)(unsigned long long)
         (__attribute__((address_space(3))) const char*)(p);
}

// Async global->LDS staging of packed f16 weights (b128 per lane, ASYNCcnt).
__device__ __forceinline__ void async_stage(const _Float16* __restrict__ g,
                                            _Float16* __restrict__ s,
                                            int halfs, int tid)
{
  unsigned base = lds_off(s);
  int n = halfs >> 3;                       // number of 16-byte chunks
  for (int i = tid; i < n; i += NTHREADS) {
    const void* ga = (const void*)(g + (i << 3));
    unsigned la = base + ((unsigned)i << 4);
    asm volatile("global_load_async_to_lds_b128 %0, %1, off"
                 :: "v"(la), "v"(ga) : "memory");
  }
}

#define WAIT_ASYNC() asm volatile("s_wait_asynccnt 0x0" ::: "memory")

// One 16-row x 64-col half-tile of a K=128 dense layer, A row from LDS.
__device__ __forceinline__ void dense_half(const _Float16* __restrict__ aRow,
                                           const _Float16* __restrict__ sW,
                                           int lane, int kh, int nh, v8f acc[4])
{
#pragma unroll
  for (int kt = 0; kt < 4; ++kt) {
    int kb = kt * 32 + kh * 16;              // A f16 layout: e0-7 -> K=kh*16+e,
    V16U A;                                  //               e8-15 -> K=kh*16+e
    A.h[0] = *(const v8h*)(aRow + kb);
    A.h[1] = *(const v8h*)(aRow + kb + 8);
#pragma unroll
    for (int nt = 0; nt < 4; ++nt) {
      const _Float16* bp = sW + (((kt << 3) + nh * 4 + nt) << 9) + lane * 16;
      V16U Bf;
      Bf.h[0] = *(const v8h*)bp;
      Bf.h[1] = *(const v8h*)(bp + 8);
      acc[nt] = __builtin_amdgcn_wmma_f32_16x16x32_f16(
          false, A.v, false, Bf.v, (short)0, acc[nt], false, false);
    }
  }
}

// ---------------------------------------------------------------------------
// Fused kernel: one workgroup per batch element b. 16 waves; wave w = (mt,nh):
//   mt = w>>1 owns pair-rows [16*mt, 16*mt+16), nh = w&1 owns 64 output cols.
// Weights double-buffered: sWa (L1, then L3), sWb (L2), async-prefetched.
// ---------------------------------------------------------------------------
__global__ __launch_bounds__(NTHREADS) void prn_fused(
    const float* __restrict__ x,        // [1024][16][128] f32
    const _Float16* __restrict__ wp,    // packed weights (65536 halfs)
    const float* __restrict__ b1, const float* __restrict__ b2,
    const float* __restrict__ b3,
    const int* __restrict__ i_idx, const int* __restrict__ j_idx,
    float* __restrict__ out)            // [1024][128] f32
{
  __shared__ __align__(16) _Float16 sWa[32768];           // 64 KB: L1 then L3
  __shared__ __align__(16) _Float16 sWb[16384];           // 32 KB: L2
  __shared__ __align__(16) _Float16 sX[16 * XS_STRIDE];   // x[b] as f16
  __shared__ __align__(16) _Float16 sH[128 * HS_STRIDE];  // activations
  __shared__ float sSum[8 * 128];                         // per-mtile partials

  const int tid  = threadIdx.x;
  const int w    = tid >> 5;
  const int mt   = w >> 1;      // M tile (0..7)
  const int nh   = w & 1;       // N half (0..1)
  const int lane = tid & 31;
  const int m    = lane & 15;
  const int kh   = lane >> 4;
  const int b    = blockIdx.x;

  // kick off async staging of layer-1 weights immediately
  async_stage(wp, sWa, 32768, tid);

  // ---- stage x[b] (2048 f32) into sX as f16, 4 contiguous elems/thread ----
  {
    const float* xb = x + (size_t)b * 2048;
    int base = tid * 4;
    int c = base >> 7, f = base & 127;
    float4 a0 = *(const float4*)(xb + base);
    v4h hv;
    hv[0] = (_Float16)a0.x; hv[1] = (_Float16)a0.y;
    hv[2] = (_Float16)a0.z; hv[3] = (_Float16)a0.w;
    *(v4h*)(&sX[c * XS_STRIDE + f]) = hv;
  }

  // pair indices for this lane's row
  const int p  = mt * 16 + m;
  const int pe = (p < 120) ? p : 0;
  const int ii = i_idx[pe];
  const int jj = j_idx[pe];

  v8f acc[4];

  WAIT_ASYNC();          // L1 weights in LDS (own wave's copies)
  __syncthreads();       // everyone's copies + sX visible

  // prefetch layer-2 weights while layer 1 computes
  async_stage(wp + 32768, sWb, 16384, tid);

  // ======================= Layer 1: K=256 (gathered pairs) ================
  {
#pragma unroll
    for (int nt = 0; nt < 4; ++nt) acc[nt] = {0.f,0.f,0.f,0.f,0.f,0.f,0.f,0.f};

    const _Float16* rowi = &sX[ii * XS_STRIDE];
    const _Float16* rowj = &sX[jj * XS_STRIDE];
#pragma unroll
    for (int kt = 0; kt < 8; ++kt) {
      const _Float16* src = (kt < 4) ? rowi : rowj;   // concat(x_i, x_j)
      int kb = (kt & 3) * 32 + kh * 16;
      V16U A;
      A.h[0] = *(const v8h*)(src + kb);
      A.h[1] = *(const v8h*)(src + kb + 8);
#pragma unroll
      for (int nt = 0; nt < 4; ++nt) {
        const _Float16* bp = sWa + (((kt << 3) + nh * 4 + nt) << 9) + lane * 16;
        V16U Bf;
        Bf.h[0] = *(const v8h*)bp;
        Bf.h[1] = *(const v8h*)(bp + 8);
        acc[nt] = __builtin_amdgcn_wmma_f32_16x16x32_f16(
            false, A.v, false, Bf.v, (short)0, acc[nt], false, false);
      }
    }
    // bias + relu -> f16 activations
#pragma unroll
    for (int nt = 0; nt < 4; ++nt) {
      int col = (nh * 4 + nt) * 16 + m;
      float bv = b1[col];
#pragma unroll
      for (int r = 0; r < 8; ++r) {
        float vf = acc[nt][r] + bv;
        vf = vf > 0.f ? vf : 0.f;
        sH[(mt * 16 + 8 * kh + r) * HS_STRIDE + col] = (_Float16)vf;
      }
    }
  }
  WAIT_ASYNC();          // L2 weights arrived
  __syncthreads();       // L1 reads of sWa done; sH complete

  // prefetch layer-3 weights (into sWa) while layer 2 computes
  async_stage(wp + 49152, sWa, 16384, tid);

  // ======================= Layer 2: K=128 =================================
  {
#pragma unroll
    for (int nt = 0; nt < 4; ++nt) acc[nt] = {0.f,0.f,0.f,0.f,0.f,0.f,0.f,0.f};
    dense_half(&sH[(mt * 16 + m) * HS_STRIDE], sWb, lane, kh, nh, acc);
#pragma unroll
    for (int nt = 0; nt < 4; ++nt) {
      int col = (nh * 4 + nt) * 16 + m;
      float bv = b2[col];
#pragma unroll
      for (int r = 0; r < 8; ++r) {
        float vf = acc[nt][r] + bv;
        vf = vf > 0.f ? vf : 0.f;
        sH[(mt * 16 + 8 * kh + r) * HS_STRIDE + col] = (_Float16)vf;
      }
    }
  }
  WAIT_ASYNC();          // L3 weights arrived
  __syncthreads();       // L2 done; sH complete

  // ======================= Layer 3: K=128 + masked mean ===================
  {
#pragma unroll
    for (int nt = 0; nt < 4; ++nt) acc[nt] = {0.f,0.f,0.f,0.f,0.f,0.f,0.f,0.f};
    dense_half(&sH[(mt * 16 + m) * HS_STRIDE], sWa, lane, kh, nh, acc);
#pragma unroll
    for (int nt = 0; nt < 4; ++nt) {
      int col = (nh * 4 + nt) * 16 + m;
      float bv = b3[col];
      float s = 0.f;
#pragma unroll
      for (int r = 0; r < 8; ++r) {
        int row = mt * 16 + 8 * kh + r;
        float vf = acc[nt][r] + bv;
        vf = vf > 0.f ? vf : 0.f;
        if (row < 120) s += vf;        // exclude padding rows from the mean
      }
      s += __shfl_xor(s, 16, 32);      // merge the two 8-row halves
      if (kh == 0) sSum[mt * 128 + col] = s;
    }
  }
  __syncthreads();

  if (tid < 128) {
    float t = 0.f;
#pragma unroll
    for (int q = 0; q < 8; ++q) t += sSum[q * 128 + tid];
    out[(size_t)b * 128 + tid] = t * (1.0f / 120.0f);
  }
}

// ---------------------------------------------------------------------------
extern "C" void kernel_launch(void* const* d_in, const int* in_sizes, int n_in,
                              void* d_out, int out_size, void* d_ws, size_t ws_size,
                              hipStream_t stream) {
  const float* x  = (const float*)d_in[0];
  const float* W1 = (const float*)d_in[1];
  const float* b1 = (const float*)d_in[2];
  const float* W2 = (const float*)d_in[3];
  const float* b2 = (const float*)d_in[4];
  const float* W3 = (const float*)d_in[5];
  const float* b3 = (const float*)d_in[6];
  const int* ii   = (const int*)d_in[7];
  const int* jj   = (const int*)d_in[8];
  _Float16* wp    = (_Float16*)d_ws;   // 65536 halfs = 128 KB scratch

  prn_pack_weights<<<256, 256, 0, stream>>>(W1, W2, W3, wp);
  prn_fused<<<1024, NTHREADS, 0, stream>>>(x, wp, b1, b2, b3, ii, jj,
                                           (float*)d_out);
}
